// PointerGeneratorNetwork_18296560681525
// MI455X (gfx1250) — matrix operand order repaired and checked by reading, not compiled, not run on hardware
//
#include <hip/hip_runtime.h>
#include <math.h>

typedef __attribute__((ext_vector_type(8)))  float  v8f;
typedef __attribute__((ext_vector_type(8)))  __bf16 v8bf;
typedef __attribute__((ext_vector_type(16))) __bf16 v16bf;

#define VV 50000
#define BB 16
#define SS 400
#define TT 100
#define EE 128
#define HH 256

__device__ __forceinline__ float sigf(float x) { return 1.f / (1.f + expf(-x)); }

// ---------------------------------------------------------------------------
// fp32 -> bf16 conversion (grid-stride)
// ---------------------------------------------------------------------------
__global__ void to_bf16(const float* __restrict__ src, __bf16* __restrict__ dst, int n) {
    int i = blockIdx.x * blockDim.x + threadIdx.x;
    int stride = gridDim.x * blockDim.x;
    for (; i < n; i += stride) dst[i] = (__bf16)src[i];
}

__global__ void vec_add(const float* __restrict__ a, const float* __restrict__ b,
                        float* __restrict__ c, int n) {
    int i = blockIdx.x * blockDim.x + threadIdx.x;
    if (i < n) c[i] = a[i] + b[i];
}

// ---------------------------------------------------------------------------
// Encoder embedding gather -> bf16 activations x[S*B, E]
// ---------------------------------------------------------------------------
__global__ void enc_emb(const int* __restrict__ texts, const float* __restrict__ emb,
                        __bf16* __restrict__ x_bf) {
    int idx = blockIdx.x * 256 + threadIdx.x;           // S*B*E = 819200
    if (idx >= SS * BB * EE) return;
    int sb = idx >> 7, e = idx & 127;
    int tok = texts[sb];
    x_bf[(size_t)sb * EE + e] = (__bf16)emb[(size_t)tok * EE + e];
}

// Decoder embedding gather into dec_in_bf[:, 512:640]
__global__ void dec_emb(const int* __restrict__ summaries, const float* __restrict__ emb,
                        __bf16* __restrict__ dec_in_bf, int t) {
    int idx = blockIdx.x * 256 + threadIdx.x;           // B*E = 2048
    if (idx >= BB * EE) return;
    int b = idx >> 7, e = idx & 127;
    int tok = summaries[t * BB + b];
    dec_in_bf[b * 640 + 512 + e] = (__bf16)emb[(size_t)tok * EE + e];
}

__global__ void init_state(float* __restrict__ coverage, __bf16* __restrict__ dec_in_bf) {
    int i = blockIdx.x * 256 + threadIdx.x;
    if (i < SS * BB) coverage[i] = 0.f;
    if (i < BB * 640) dec_in_bf[i] = (__bf16)0.f;
}

// ---------------------------------------------------------------------------
// Generic WMMA GEMM: C[M,N] = act( A[M,K] @ W[N,K]^T + bias + addm )
// One wave per 16x16 tile.  act: 0=none, 1=relu, 2=lstm-gates (H=256 blocks)
// ---------------------------------------------------------------------------
__global__ __launch_bounds__(32) void gemm_bf16(
    const __bf16* __restrict__ A, const __bf16* __restrict__ W,
    const float* __restrict__ bias, const float* __restrict__ addm,
    float* __restrict__ C, __bf16* __restrict__ Cb,
    int K, int N, int act)
{
    int n0 = blockIdx.x * 16, m0 = blockIdx.y * 16;
    int lane = threadIdx.x & 31;
    int nl = lane & 15, half = lane >> 4;
    const __bf16* arow = A + (size_t)(m0 + nl) * K;   // A row for this lane
    const __bf16* wrow = W + (size_t)(n0 + nl) * K;   // W row (= B column) for this lane
    v8f acc = {};
    for (int k0 = 0; k0 < K; k0 += 32) {
        v8bf a0 = *(const v8bf*)(arow + k0 + 16 * half);
        v8bf a1 = *(const v8bf*)(arow + k0 + 16 * half + 8);
        v16bf a;
#pragma unroll
        for (int i = 0; i < 8; ++i) { a[i] = a0[i]; a[8 + i] = a1[i]; }
        v16bf bv = *(const v16bf*)(wrow + k0 + 16 * half);
        acc = __builtin_amdgcn_wmma_f32_16x16x32_bf16(false, a, false, bv,
                                                      (short)0, acc, false, false);
    }
    int n_g = n0 + nl;
    float bsv = bias ? bias[n_g] : 0.f;
#pragma unroll
    for (int i = 0; i < 8; ++i) {
        int mr = m0 + i + 8 * half;
        size_t o = (size_t)mr * N + n_g;
        float v = acc[i] + bsv;
        if (addm) v += addm[o];
        if (act == 1) v = fmaxf(v, 0.f);
        else if (act == 2) {
            if (((n_g >> 8) & 3) == 2) v = tanhf(v);   // g-gate block
            else v = sigf(v);                           // i, f, o gates
        }
        if (C)  C[o]  = v;
        if (Cb) Cb[o] = (__bf16)v;
    }
}

// ---------------------------------------------------------------------------
// Persistent bidirectional LSTM scan.  grid = 2 blocks (dir), 1024 threads.
// X* = precomputed x@Wih^T + bih, raw.  h/c kept in LDS; WMMA recurrent GEMM.
// ---------------------------------------------------------------------------
__global__ __launch_bounds__(1024) void enc_scan(
    const float* __restrict__ Xf, const float* __restrict__ Xb,
    const __bf16* __restrict__ WhhF, const __bf16* __restrict__ WhhB,
    const float* __restrict__ bhhF, const float* __restrict__ bhhB,
    float* __restrict__ enc_out, __bf16* __restrict__ enc_out_bf,
    __bf16* __restrict__ hcat_bf, __bf16* __restrict__ ccat_bf)
{
    __shared__ __align__(16) __bf16 h_bf[16][256];
    __shared__ float c_st[16][256];
    __shared__ float g[16][1024];

    int dir = blockIdx.x;
    const float*  X    = dir ? Xb   : Xf;
    const __bf16* Whh  = dir ? WhhB : WhhF;
    const float*  bhh  = dir ? bhhB : bhhF;

    int tid = threadIdx.x;
    for (int i = tid; i < 16 * 256; i += 1024) {
        (&h_bf[0][0])[i] = (__bf16)0.f;
        (&c_st[0][0])[i] = 0.f;
    }
    __syncthreads();

    int wid = tid >> 5, lane = tid & 31;
    int nl = lane & 15, half = lane >> 4;
    int n_g0 = wid * 16 + nl;
    int n_g1 = (wid + 32) * 16 + nl;
    const __bf16* w0 = Whh + (size_t)n_g0 * 256;
    const __bf16* w1 = Whh + (size_t)n_g1 * 256;

    for (int t = 0; t < SS; ++t) {
        int te = dir ? (SS - 1 - t) : t;
        // ---- recurrent GEMM: g = h @ Whh^T (two 16x16 tiles per wave) ----
        v8f acc0 = {}, acc1 = {};
        for (int k0 = 0; k0 < 256; k0 += 32) {
            v8bf a0 = *(const v8bf*)&h_bf[nl][k0 + 16 * half];
            v8bf a1 = *(const v8bf*)&h_bf[nl][k0 + 16 * half + 8];
            v16bf a;
#pragma unroll
            for (int i = 0; i < 8; ++i) { a[i] = a0[i]; a[8 + i] = a1[i]; }
            v16bf b0 = *(const v16bf*)(w0 + k0 + 16 * half);
            v16bf b1 = *(const v16bf*)(w1 + k0 + 16 * half);
            acc0 = __builtin_amdgcn_wmma_f32_16x16x32_bf16(false, a, false, b0,
                                                           (short)0, acc0, false, false);
            acc1 = __builtin_amdgcn_wmma_f32_16x16x32_bf16(false, a, false, b1,
                                                           (short)0, acc1, false, false);
        }
#pragma unroll
        for (int i = 0; i < 8; ++i) {
            int mr = i + 8 * half;
            const float* xr = X + ((size_t)te * BB + mr) * 1024;
            g[mr][n_g0] = acc0[i] + xr[n_g0] + bhh[n_g0];
            g[mr][n_g1] = acc1[i] + xr[n_g1] + bhh[n_g1];
        }
        __syncthreads();
        // ---- elementwise gate update ----
        for (int r = 0; r < 4; ++r) {
            int idx = tid + r * 1024;
            int b = idx >> 8, jj = idx & 255;
            float ig = sigf(g[b][jj]);
            float fg = sigf(g[b][256 + jj]);
            float gg = tanhf(g[b][512 + jj]);
            float og = sigf(g[b][768 + jj]);
            float cn = fg * c_st[b][jj] + ig * gg;
            float hn = og * tanhf(cn);
            c_st[b][jj] = cn;
            h_bf[b][jj] = (__bf16)hn;
            size_t o = ((size_t)te * BB + b) * 512 + dir * 256 + jj;
            enc_out[o]    = hn;
            enc_out_bf[o] = (__bf16)hn;
        }
        __syncthreads();
    }
    // final states -> concat buffers for the reduce GEMMs
    for (int idx = tid; idx < 16 * 256; idx += 1024) {
        int b = idx >> 8, jj = idx & 255;
        hcat_bf[b * 512 + dir * 256 + jj] = h_bf[b][jj];
        ccat_bf[b * 512 + dir * 256 + jj] = (__bf16)c_st[b][jj];
    }
}

// ---------------------------------------------------------------------------
// Decoder LSTM elementwise (gates already sigmoid/tanh'd by GEMM epilogue)
// ---------------------------------------------------------------------------
__global__ void dec_lstm_elem(const float* __restrict__ gact, const float* __restrict__ c0d,
                              __bf16* __restrict__ hidden_bf, __bf16* __restrict__ outcat_bf) {
    int idx = blockIdx.x * 256 + threadIdx.x;   // B*H = 4096
    int b = idx >> 8, jj = idx & 255;
    float ig = gact[b * 1024 + jj];
    float fg = gact[b * 1024 + 256 + jj];
    float gg = gact[b * 1024 + 512 + jj];
    float og = gact[b * 1024 + 768 + jj];
    float cn = fg * c0d[b * 256 + jj] + ig * gg;
    float hh = og * tanhf(cn);
    hidden_bf[b * 512 + jj]        = (__bf16)hh;
    hidden_bf[b * 512 + 256 + jj]  = (__bf16)cn;
    outcat_bf[b * 768 + jj]        = (__bf16)hh;
}

// ---------------------------------------------------------------------------
// Fused attention: e -> softmax over S -> coverage update -> context
// grid = B blocks, 256 threads
// ---------------------------------------------------------------------------
__global__ __launch_bounds__(256) void attn_kernel(
    const float* __restrict__ enc_feat, const float* __restrict__ enc_out,
    const float* __restrict__ dec_f,
    const float* __restrict__ Wcov, const float* __restrict__ bcov,
    const float* __restrict__ Wattn, const float* __restrict__ battn,
    float* __restrict__ coverage, __bf16* __restrict__ dec_in_bf,
    __bf16* __restrict__ outcat_bf,
    float* __restrict__ attns_out, float* __restrict__ covs_out, int t)
{
    int b = blockIdx.x, tid = threadIdx.x;
    __shared__ float e_l[SS];
    __shared__ float df_l[512], wc_l[512], bc_l[512], wa_l[512];
    __shared__ float red[256];
    for (int k = tid; k < 512; k += 256) {
        df_l[k] = dec_f[b * 512 + k];
        wc_l[k] = Wcov[k]; bc_l[k] = bcov[k]; wa_l[k] = Wattn[k];
    }
    __syncthreads();
    int wid = tid >> 5, lane = tid & 31;
    float bt = battn[0];
    for (int s = wid; s < SS; s += 8) {
        float cs = coverage[s * BB + b];
        const float* ef = enc_feat + ((size_t)s * BB + b) * 512;
        float part = 0.f;
        for (int k = lane; k < 512; k += 32) {
            float v = tanhf(ef[k] + df_l[k] + cs * wc_l[k] + bc_l[k]);
            part += v * wa_l[k];
        }
        for (int o = 16; o; o >>= 1) part += __shfl_xor(part, o, 32);
        if (lane == 0) e_l[s] = part + bt;
    }
    __syncthreads();
    // softmax over S
    float mx = -1e30f;
    for (int s = tid; s < SS; s += 256) mx = fmaxf(mx, e_l[s]);
    red[tid] = mx; __syncthreads();
    for (int o = 128; o; o >>= 1) { if (tid < o) red[tid] = fmaxf(red[tid], red[tid + o]); __syncthreads(); }
    mx = red[0]; __syncthreads();
    float sm = 0.f;
    for (int s = tid; s < SS; s += 256) { float e = expf(e_l[s] - mx); e_l[s] = e; sm += e; }
    red[tid] = sm; __syncthreads();
    for (int o = 128; o; o >>= 1) { if (tid < o) red[tid] += red[tid + o]; __syncthreads(); }
    float inv = 1.f / red[0];
    for (int s = tid; s < SS; s += 256) {
        float a = e_l[s] * inv; e_l[s] = a;
        float cn = coverage[s * BB + b] + a;
        coverage[s * BB + b] = cn;
        attns_out[((size_t)t * SS + s) * BB + b] = a;
        covs_out [((size_t)t * SS + s) * BB + b] = cn;
    }
    __syncthreads();
    // context = sum_s a[s] * enc_out[s,b,:]
    for (int h2 = tid; h2 < 512; h2 += 256) {
        float acc = 0.f;
        const float* eo = enc_out + (size_t)b * 512 + h2;
        for (int s = 0; s < SS; ++s) acc += e_l[s] * eo[(size_t)s * BB * 512];
        dec_in_bf[b * 640 + h2]        = (__bf16)acc;   // carry for next step
        outcat_bf[b * 768 + 256 + h2]  = (__bf16)acc;   // for logits1 this step
    }
}

// ---------------------------------------------------------------------------
// In-place softmax over V per batch row.  grid = B blocks, 256 threads.
// ---------------------------------------------------------------------------
__global__ __launch_bounds__(256) void softmax_v(float* __restrict__ outs_t) {
    int b = blockIdx.x, tid = threadIdx.x;
    float* row = outs_t + (size_t)b * VV;
    __shared__ float red[256];
    float mx = -1e30f;
    for (int i = tid; i < VV; i += 256) mx = fmaxf(mx, row[i]);
    red[tid] = mx; __syncthreads();
    for (int o = 128; o; o >>= 1) { if (tid < o) red[tid] = fmaxf(red[tid], red[tid + o]); __syncthreads(); }
    mx = red[0]; __syncthreads();
    float sm = 0.f;
    for (int i = tid; i < VV; i += 256) { float e = expf(row[i] - mx); row[i] = e; sm += e; }
    red[tid] = sm; __syncthreads();
    for (int o = 128; o; o >>= 1) { if (tid < o) red[tid] += red[tid + o]; __syncthreads(); }
    float inv = 1.f / red[0];
    for (int i = tid; i < VV; i += 256) row[i] *= inv;
}

// ---------------------------------------------------------------------------
extern "C" void kernel_launch(void* const* d_in, const int* in_sizes, int n_in,
                              void* d_out, int out_size, void* d_ws, size_t ws_size,
                              hipStream_t stream) {
    (void)in_sizes; (void)n_in; (void)out_size; (void)ws_size;
    const int*   texts     = (const int*)d_in[0];
    const int*   summaries = (const int*)d_in[1];
    const float* emb_enc = (const float*)d_in[2];
    const float* Wih_f = (const float*)d_in[3];  const float* Whh_f = (const float*)d_in[4];
    const float* bih_f = (const float*)d_in[5];  const float* bhh_f = (const float*)d_in[6];
    const float* Wih_b = (const float*)d_in[7];  const float* Whh_b = (const float*)d_in[8];
    const float* bih_b = (const float*)d_in[9];  const float* bhh_b = (const float*)d_in[10];
    const float* Wlin = (const float*)d_in[11];  const float* blin = (const float*)d_in[12];
    const float* Wrh  = (const float*)d_in[13];  const float* brh  = (const float*)d_in[14];
    const float* Wrc  = (const float*)d_in[15];  const float* brc  = (const float*)d_in[16];
    const float* emb_dec = (const float*)d_in[17];
    const float* Wih_d = (const float*)d_in[18]; const float* Whh_d = (const float*)d_in[19];
    const float* bih_d = (const float*)d_in[20]; const float* bhh_d = (const float*)d_in[21];
    const float* Wctx = (const float*)d_in[22];  const float* bctx = (const float*)d_in[23];
    const float* Wout1 = (const float*)d_in[24]; const float* bout1 = (const float*)d_in[25];
    const float* Wout2 = (const float*)d_in[26]; const float* bout2 = (const float*)d_in[27];
    const float* Wfeat = (const float*)d_in[28]; const float* bfeat = (const float*)d_in[29];
    const float* Wcov = (const float*)d_in[30];  const float* bcov = (const float*)d_in[31];
    const float* Wattn = (const float*)d_in[32]; const float* battn = (const float*)d_in[33];

    float* outF = (float*)d_out;
    float* attns_out = outF + (size_t)TT * BB * VV;
    float* covs_out  = attns_out + (size_t)TT * SS * BB;

    // ---- workspace layout ----
    char* ws = (char*)d_ws;
    size_t off = 0;
    auto alloc = [&](size_t bytes) -> char* {
        char* p = ws + off;
        off += (bytes + 255) & ~(size_t)255;
        return p;
    };
    __bf16* wbf_ih_f = (__bf16*)alloc(1024 * 128 * 2);
    __bf16* wbf_hh_f = (__bf16*)alloc(1024 * 256 * 2);
    __bf16* wbf_ih_b = (__bf16*)alloc(1024 * 128 * 2);
    __bf16* wbf_hh_b = (__bf16*)alloc(1024 * 256 * 2);
    __bf16* wbf_lin  = (__bf16*)alloc(512 * 512 * 2);
    __bf16* wbf_rh   = (__bf16*)alloc(256 * 512 * 2);
    __bf16* wbf_rc   = (__bf16*)alloc(256 * 512 * 2);
    __bf16* wbf_ih_d = (__bf16*)alloc(1024 * 128 * 2);
    __bf16* wbf_hh_d = (__bf16*)alloc(1024 * 256 * 2);
    __bf16* wbf_ctx  = (__bf16*)alloc(128 * 640 * 2);
    __bf16* wbf_out1 = (__bf16*)alloc(256 * 768 * 2);
    __bf16* wbf_out2 = (__bf16*)alloc((size_t)VV * 256 * 2);
    __bf16* wbf_feat = (__bf16*)alloc(512 * 512 * 2);
    __bf16* x_bf     = (__bf16*)alloc((size_t)SS * BB * EE * 2);
    float*  Xf       = (float*) alloc((size_t)SS * BB * 1024 * 4);
    float*  Xb       = (float*) alloc((size_t)SS * BB * 1024 * 4);
    float*  enc_out  = (float*) alloc((size_t)SS * BB * 512 * 4);
    __bf16* enc_out_bf = (__bf16*)alloc((size_t)SS * BB * 512 * 2);
    float*  enc_feat = (float*) alloc((size_t)SS * BB * 512 * 4);
    __bf16* hcat_bf  = (__bf16*)alloc(16 * 512 * 2);
    __bf16* ccat_bf  = (__bf16*)alloc(16 * 512 * 2);
    __bf16* h0d_bf   = (__bf16*)alloc(16 * 256 * 2);
    float*  c0d      = (float*) alloc(16 * 256 * 4);
    float*  bsum     = (float*) alloc(1024 * 4);
    float*  hterm    = (float*) alloc(16 * 1024 * 4);
    __bf16* dec_in_bf  = (__bf16*)alloc(16 * 640 * 2);
    __bf16* xt_bf      = (__bf16*)alloc(16 * 128 * 2);
    float*  gact       = (float*) alloc(16 * 1024 * 4);
    __bf16* hidden_bf  = (__bf16*)alloc(16 * 512 * 2);
    __bf16* outcat_bf  = (__bf16*)alloc(16 * 768 * 2);
    float*  dec_f      = (float*) alloc(16 * 512 * 4);
    float*  coverage   = (float*) alloc((size_t)SS * BB * 4);
    __bf16* logits1_bf = (__bf16*)alloc(16 * 256 * 2);

    auto conv = [&](const float* src, __bf16* dst, int n) {
        int blocks = (n + 1023) / 1024;
        to_bf16<<<blocks, 256, 0, stream>>>(src, dst, n);
    };
    conv(Wih_f, wbf_ih_f, 1024 * 128);
    conv(Whh_f, wbf_hh_f, 1024 * 256);
    conv(Wih_b, wbf_ih_b, 1024 * 128);
    conv(Whh_b, wbf_hh_b, 1024 * 256);
    conv(Wlin,  wbf_lin,  512 * 512);
    conv(Wrh,   wbf_rh,   256 * 512);
    conv(Wrc,   wbf_rc,   256 * 512);
    conv(Wih_d, wbf_ih_d, 1024 * 128);
    conv(Whh_d, wbf_hh_d, 1024 * 256);
    conv(Wctx,  wbf_ctx,  128 * 640);
    conv(Wout1, wbf_out1, 256 * 768);
    conv(Wout2, wbf_out2, VV * 256);
    conv(Wfeat, wbf_feat, 512 * 512);

    // encoder embedding + input transforms
    enc_emb<<<(SS * BB * EE + 255) / 256, 256, 0, stream>>>(texts, emb_enc, x_bf);
    gemm_bf16<<<dim3(1024 / 16, SS * BB / 16), 32, 0, stream>>>(
        x_bf, wbf_ih_f, bih_f, nullptr, Xf, nullptr, 128, 1024, 0);
    gemm_bf16<<<dim3(1024 / 16, SS * BB / 16), 32, 0, stream>>>(
        x_bf, wbf_ih_b, bih_b, nullptr, Xb, nullptr, 128, 1024, 0);

    // persistent bidirectional LSTM scan
    enc_scan<<<2, 1024, 0, stream>>>(Xf, Xb, wbf_hh_f, wbf_hh_b, bhh_f, bhh_b,
                                     enc_out, enc_out_bf, hcat_bf, ccat_bf);

    // enc_feat = enc_out @ Wlin^T + blin
    gemm_bf16<<<dim3(512 / 16, SS * BB / 16), 32, 0, stream>>>(
        enc_out_bf, wbf_lin, blin, nullptr, enc_feat, nullptr, 512, 512, 0);

    // reduce states: h0d = relu(hcat@Wrh^T+brh), c0d = relu(ccat@Wrc^T+brc)
    gemm_bf16<<<dim3(256 / 16, 1), 32, 0, stream>>>(
        hcat_bf, wbf_rh, brh, nullptr, nullptr, h0d_bf, 512, 256, 1);
    gemm_bf16<<<dim3(256 / 16, 1), 32, 0, stream>>>(
        ccat_bf, wbf_rc, brc, nullptr, c0d, nullptr, 512, 256, 1);

    // hterm = h0d @ Whh_d^T + (bih_d + bhh_d)   (constant across decoder steps)
    vec_add<<<4, 256, 0, stream>>>(bih_d, bhh_d, bsum, 1024);
    gemm_bf16<<<dim3(1024 / 16, 1), 32, 0, stream>>>(
        h0d_bf, wbf_hh_d, bsum, nullptr, hterm, nullptr, 256, 1024, 0);

    init_state<<<(BB * 640 + 255) / 256, 256, 0, stream>>>(coverage, dec_in_bf);

    // ---- decoder ----
    for (int t = 0; t < TT; ++t) {
        float* outs_t = outF + (size_t)t * BB * VV;
        dec_emb<<<(BB * EE + 255) / 256, 256, 0, stream>>>(summaries, emb_dec, dec_in_bf, t);
        // xt = [context, emb] @ Wctx^T + bctx
        gemm_bf16<<<dim3(128 / 16, 1), 32, 0, stream>>>(
            dec_in_bf, wbf_ctx, bctx, nullptr, nullptr, xt_bf, 640, 128, 0);
        // gates = act(xt @ Wih_d^T + hterm)
        gemm_bf16<<<dim3(1024 / 16, 1), 32, 0, stream>>>(
            xt_bf, wbf_ih_d, nullptr, hterm, gact, nullptr, 128, 1024, 2);
        dec_lstm_elem<<<16, 256, 0, stream>>>(gact, c0d, hidden_bf, outcat_bf);
        // dec_f = [hh,c] @ Wfeat^T + bfeat
        gemm_bf16<<<dim3(512 / 16, 1), 32, 0, stream>>>(
            hidden_bf, wbf_feat, bfeat, nullptr, dec_f, nullptr, 512, 512, 0);
        attn_kernel<<<BB, 256, 0, stream>>>(enc_feat, enc_out, dec_f,
                                            Wcov, bcov, Wattn, battn,
                                            coverage, dec_in_bf, outcat_bf,
                                            attns_out, covs_out, t);
        // logits1 = [hh,ctx] @ Wout1^T + bout1
        gemm_bf16<<<dim3(256 / 16, 1), 32, 0, stream>>>(
            outcat_bf, wbf_out1, bout1, nullptr, nullptr, logits1_bf, 768, 256, 0);
        // logits2 = logits1 @ Wout2^T + bout2  (L2-resident 50000x256 weight)
        gemm_bf16<<<dim3(VV / 16, 1), 32, 0, stream>>>(
            logits1_bf, wbf_out2, bout2, nullptr, outs_t, nullptr, 256, VV, 0);
        softmax_v<<<BB, 256, 0, stream>>>(outs_t);
    }
}
